// RINet_attention_17145509445683
// MI455X (gfx1250) — matrix operand order repaired and checked by reading, not compiled, not run on hardware
//
#include <hip/hip_runtime.h>
#include <hip/hip_bf16.h>
#include <math.h>

#define BT 128
#define NWAVES 4
#define EPSV 1e-5f

typedef __attribute__((ext_vector_type(16))) _Float16 v16h;
typedef __attribute__((ext_vector_type(8)))  float    v8f;
typedef __attribute__((ext_vector_type(4))) unsigned  u32x4;
typedef __attribute__((ext_vector_type(4))) int       i32x4;
typedef __attribute__((ext_vector_type(8))) int       i32x8;

union AFrag { v16h v; _Float16 e[16]; };
union CFrag { v8f  v; float     e[8];  };

struct Params {
    const float* x; const float* y;
    const float* attW[8]; const float* attb[8];
    const float* convW[7]; const float* convb[7];
    const float* bng[7]; const float* bnb[7]; const float* bnm[7]; const float* bnv[7];
    const float* linW1; const float* linb1; const float* linW2; const float* linb2;
};

__device__ __forceinline__ float sigmoid_f(float x) { return 1.0f / (1.0f + __expf(-x)); }
__device__ __forceinline__ int   imin(int a, int b) { return a < b ? a : b; }
__device__ __forceinline__ unsigned rfl(unsigned x) { return (unsigned)__builtin_amdgcn_readfirstlane((int)x); }

// ---------------------------------------------------------------------------
// TDM: async 1-D tile load (nelem f32, contiguous) from global to LDS.
// Issued by one wave; completion via s_wait_tensorcnt. D# per ISA ch.8:
//   group0: count=1 | lds_addr | global_addr[56:0] | type=2
//   group1: data_size=4B | tensor_dim0=tile_dim0=nelem | dim1=1 | strides
// ---------------------------------------------------------------------------
__device__ __forceinline__
void tdm_load_f32(const float* gsrc, float* ldsdst, int nelem)
{
    const unsigned lds = rfl((unsigned)(size_t)ldsdst);          // low 32 = LDS offset
    const unsigned long long ga = (unsigned long long)(size_t)gsrc;
    const unsigned galo = rfl((unsigned)ga);
    const unsigned gahi = rfl(((unsigned)(ga >> 32) & 0x01FFFFFFu) | (2u << 30)); // type=2
    const unsigned ne   = rfl((unsigned)nelem);

    const unsigned d0 = 1u;                                       // count=1, user D#
    const unsigned e0 = 2u << 16;                                 // data_size = 4 bytes
    const unsigned e1 = (ne & 0xFFFFu) << 16;                     // tensor_dim0 lo16
    const unsigned e2 = (ne >> 16) | (1u << 16);                  // dim0 hi16 | tensor_dim1=1
    const unsigned e3 = (ne & 0xFFFFu) << 16;                     // tile_dim0
    const unsigned e4 = 1u;                                       // tile_dim1=1
    const unsigned e5 = ne;                                       // dim0_stride lo32
    const unsigned e6 = (ne & 0xFFFFu) << 16;                     // dim1_stride lo16
    const unsigned e7 = 0u;

    u32x4 g0 = { d0, lds, galo, gahi };
    i32x8 g1 = { (int)e0, (int)e1, (int)e2, (int)e3, (int)e4, (int)e5, (int)e6, (int)e7 };
#if __has_builtin(__builtin_amdgcn_tensor_load_to_lds)
    i32x4 z4 = { 0, 0, 0, 0 };
  #if __clang_major__ >= 23
    i32x8 z8 = { 0, 0, 0, 0, 0, 0, 0, 0 };
    __builtin_amdgcn_tensor_load_to_lds(g0, g1, z4, z4, z8, 0);
  #else
    __builtin_amdgcn_tensor_load_to_lds(g0, g1, z4, z4, 0);
  #endif
#else
    asm volatile("tensor_load_to_lds %0, %1" :: "s"(g0), "s"(g1) : "memory");
#endif
}

// ---------------------------------------------------------------------------
// SE gate: h *= sigmoid(W @ mean_L(h) + b); optionally emit gate*mean as the
// pooled feature (valid because the gate is channel-uniform over L).
// ---------------------------------------------------------------------------
__device__ __forceinline__
void se_att(float* h, const float* __restrict__ W, const float* __restrict__ b,
            int C, int L, float* red, float* gate, float* featOut)
{
    const int t = threadIdx.x;
    const int n = C * L;
    for (int i = t; i < C; i += BT) red[i] = 0.0f;
    __syncthreads();
    for (int i = t; i < n; i += BT) atomicAdd(&red[i / L], h[i]);   // ds_add_f32
    __syncthreads();
    if (t < C) {
        const float invL = 1.0f / (float)L;
        float acc = b[t];
        for (int j = 0; j < C; ++j) acc += W[t * C + j] * (red[j] * invL);
        const float g = sigmoid_f(acc);
        gate[t] = g;
        if (featOut) featOut[t] = g * red[t] * invL;
    }
    __syncthreads();
    for (int i = t; i < n; i += BT) h[i] *= gate[i / L];
    __syncthreads();
}

// ---------------------------------------------------------------------------
// Stride-r phase selection by max L1 mass, then gather the winning phase.
// ---------------------------------------------------------------------------
__device__ __forceinline__
void down_sel(const float* hin, float* hout, int C, int L, int r,
              float* red, int* bestp)
{
    const int t = threadIdx.x;
    for (int i = t; i < r; i += BT) red[i] = 0.0f;
    __syncthreads();
    for (int i = t; i < C * L; i += BT) {
        int l = i % L;
        atomicAdd(&red[l % r], fabsf(hin[i]));
    }
    __syncthreads();
    if (t == 0) {
        int best = 0;
        for (int p = 1; p < r; ++p) if (red[p] > red[best]) best = p;  // first-max
        *bestp = best;
    }
    __syncthreads();
    const int Lo = L / r;
    const int best = *bestp;
    for (int i = t; i < C * Lo; i += BT) {
        int c = i / Lo, j = i - c * Lo;
        hout[c * Lo + j] = hin[c * L + j * r + best];
    }
    __syncthreads();
}

// ---------------------------------------------------------------------------
// Circular conv1d (K=3) + BN(eval) + LeakyReLU as implicit GEMM on WMMA.
// BN scale/shift precomputed per-channel into LDS (single v_rsq_f32, no
// per-tile global reloads). Only K-dim padding needs true zeros; applied as
// multiply-by-{0,1} so loads stay unconditional. Tile/k loops are
// readfirstlane-uniform: EXEC is all-ones around every v_wmma.
// ---------------------------------------------------------------------------
__device__ __forceinline__
void conv_bn_lrelu(const float* __restrict__ W,
                   const float* __restrict__ cb,
                   const float* __restrict__ g,
                   const float* __restrict__ bb,
                   const float* __restrict__ m,
                   const float* __restrict__ v,
                   const float* hin, float* hout,
                   float* scl, float* shl,
                   int Cin, int Cout, int L)
{
    const int t    = threadIdx.x;
    const int lane = t & 31;
    const int wid  = __builtin_amdgcn_readfirstlane(t >> 5);   // uniform
    const int grp  = lane >> 4;
    const int mn   = lane & 15;
    const int Kdim = 3 * Cin;
    const int ksteps = (Kdim + 31) >> 5;
    const int Mt = (Cout + 15) >> 4;
    const int Nt = (L + 15) >> 4;
    const int total = Mt * Nt;

    if (t < Cout) {                                  // fold BN once per stage
        const float s = g[t] * __builtin_amdgcn_rsqf(v[t] + EPSV);
        scl[t] = s;
        shl[t] = (cb[t] - m[t]) * s + bb[t];
    }
    __syncthreads();

    for (int tile = wid; tile < total; tile += NWAVES) {   // uniform loop
        const int mt = tile / Nt;
        const int nt = tile - mt * Nt;
        const int row_c = imin((mt << 4) + mn, Cout - 1);  // clamped A row
        const int col_c = imin((nt << 4) + mn, L - 1);     // clamped B col

        CFrag acc;
        #pragma unroll
        for (int r = 0; r < 8; ++r) acc.e[r] = 0.0f;

        for (int ks = 0; ks < ksteps; ++ks) {              // uniform loop
            AFrag a, bfr;
            #pragma unroll
            for (int vv = 0; vv < 8; ++vv) {
                const int kA = (ks << 5) + ((vv < 4) ? (vv << 1) : (16 + ((vv - 4) << 1))) + (grp << 3);
                const int kB = (ks << 5) + ((vv < 4) ? (vv << 1) : (16 + ((vv - 4) << 1))) + (grp << 3);
                #pragma unroll
                for (int h2 = 0; h2 < 2; ++h2) {
                    const int   ka    = kA + h2;
                    const int   ka_c  = imin(ka, Kdim - 1);
                    const float kmA   = (ka < Kdim) ? 1.0f : 0.0f;  // folds when Kdim%32==0
                    const float wv    = W[row_c * Kdim + ka_c];
                    a.e[2 * vv + h2]  = (_Float16)(wv * kmA);
                    const int   kb    = kB + h2;
                    const int   kb_c  = imin(kb, Kdim - 1);
                    const float kmB   = (kb < Kdim) ? 1.0f : 0.0f;
                    const int   ci    = kb_c / 3;
                    const int   kk    = kb_c - 3 * ci;
                    int l = col_c + kk; if (l >= L) l -= L;         // circular pad
                    const float hv    = hin[ci * L + l];
                    bfr.e[2 * vv + h2] = (_Float16)(hv * kmB);
                }
            }
            acc.v = __builtin_amdgcn_wmma_f32_16x16x32_f16(
                false, a.v, false, bfr.v, (short)0, acc.v, false, false);
        }

        #pragma unroll
        for (int r = 0; r < 8; ++r) {
            const int co = (mt << 4) + r + (grp << 3);
            const int l  = (nt << 4) + mn;
            if (co < Cout && l < L) {
                const float z = acc.e[r] * scl[co] + shl[co];
                hout[co * L + l] = (z >= 0.0f) ? z : 0.1f * z;
            }
        }
    }
    __syncthreads();
}

// ---------------------------------------------------------------------------
// Stage 1: one workgroup per sample; whole pipeline resident in LDS.
// Input tile arrives via the Tensor Data Mover (async, TENSORcnt-tracked).
// ---------------------------------------------------------------------------
__global__ __launch_bounds__(BT)
void features_kernel(Params P, float* __restrict__ feat_g, int B)
{
    __shared__ float hA[16 * 368];
    __shared__ float hB[16 * 368];
    __shared__ float red[160];
    __shared__ float gate[128];
    __shared__ int   bestp;

    const int s = blockIdx.x;                 // 0 .. 2B-1
    const int t = threadIdx.x;
    const int wid = __builtin_amdgcn_readfirstlane(t >> 5);
    const float* src = (s < B) ? (P.x + (size_t)s * 4320)
                               : (P.y + (size_t)(s - B) * 4320);

    if (wid == 0) tdm_load_f32(src, hA, 12 * 360);    // tensor_load_to_lds
    __builtin_amdgcn_s_wait_tensorcnt(0);             // s_wait_tensorcnt
    __syncthreads();

    float* feat = feat_g + (size_t)s * 288;

    se_att(hA, P.attW[0], P.attb[0], 12, 360, red, gate, nullptr);
    conv_bn_lrelu(P.convW[0], P.convb[0], P.bng[0], P.bnb[0], P.bnm[0], P.bnv[0], hA, hB, red, gate, 12, 12, 360);
    se_att(hB, P.attW[1], P.attb[1], 12, 360, red, gate, nullptr);
    conv_bn_lrelu(P.convW[1], P.convb[1], P.bng[1], P.bnb[1], P.bnm[1], P.bnv[1], hB, hA, red, gate, 12, 16, 360);
    se_att(hA, P.attW[2], P.attb[2], 16, 360, red, gate, feat + 0);

    down_sel(hA, hB, 16, 360, 3, red, &bestp);
    conv_bn_lrelu(P.convW[2], P.convb[2], P.bng[2], P.bnb[2], P.bnm[2], P.bnv[2], hB, hA, red, gate, 16, 16, 120);
    se_att(hA, P.attW[3], P.attb[3], 16, 120, red, gate, feat + 16);

    down_sel(hA, hB, 16, 120, 3, red, &bestp);
    conv_bn_lrelu(P.convW[3], P.convb[3], P.bng[3], P.bnb[3], P.bnm[3], P.bnv[3], hB, hA, red, gate, 16, 32, 40);
    se_att(hA, P.attW[4], P.attb[4], 32, 40, red, gate, feat + 32);

    down_sel(hA, hB, 32, 40, 2, red, &bestp);
    conv_bn_lrelu(P.convW[4], P.convb[4], P.bng[4], P.bnb[4], P.bnm[4], P.bnv[4], hB, hA, red, gate, 32, 32, 20);
    se_att(hA, P.attW[5], P.attb[5], 32, 20, red, gate, feat + 64);

    down_sel(hA, hB, 32, 20, 2, red, &bestp);
    conv_bn_lrelu(P.convW[5], P.convb[5], P.bng[5], P.bnb[5], P.bnm[5], P.bnv[5], hB, hA, red, gate, 32, 64, 10);
    se_att(hA, P.attW[6], P.attb[6], 64, 10, red, gate, feat + 96);

    down_sel(hA, hB, 64, 10, 2, red, &bestp);
    conv_bn_lrelu(P.convW[6], P.convb[6], P.bng[6], P.bnb[6], P.bnm[6], P.bnv[6], hB, hA, red, gate, 64, 128, 5);
    se_att(hA, P.attW[7], P.attb[7], 128, 5, red, gate, feat + 160);
}

// ---------------------------------------------------------------------------
// Stage 2: head. 16 pairs per block. D[16x128] = |f_x - f_y|[16x288] x W1^T,
// K = 288 = 9 WMMA k-steps exactly (maskless). Then leaky, 128->1 dot,
// sigmoid, and the L2 norm of the diff.
// ---------------------------------------------------------------------------
__global__ __launch_bounds__(BT)
void head_kernel(Params P, const float* __restrict__ feat, float* __restrict__ out, int B)
{
    __shared__ float    d32[16 * 288];
    __shared__ _Float16 d16[16 * 288];
    __shared__ float    hmid[16 * 128];

    const int t    = threadIdx.x;
    const int lane = t & 31;
    const int wid  = __builtin_amdgcn_readfirstlane(t >> 5);   // uniform
    const int grp  = lane >> 4;
    const int mn   = lane & 15;
    const int p0   = blockIdx.x * 16;

    for (int i = t; i < 16 * 288; i += BT) {
        int pr = i / 288, k = i - pr * 288;
        int pair = imin(p0 + pr, B - 1);
        float dv = fabsf(feat[(size_t)pair * 288 + k] - feat[(size_t)(pair + B) * 288 + k]);
        d32[i] = dv;
        d16[i] = (_Float16)dv;
    }
    __syncthreads();

    for (int nt = wid; nt < 8; nt += NWAVES) {      // uniform: 8 n-tiles, 2/wave
        CFrag acc;
        #pragma unroll
        for (int r = 0; r < 8; ++r) acc.e[r] = 0.0f;
        const int ng = (nt << 4) + mn;              // out neuron 0..127
        for (int ks = 0; ks < 9; ++ks) {
            AFrag a, bfr;
            #pragma unroll
            for (int vv = 0; vv < 8; ++vv) {
                const int kA = (ks << 5) + ((vv < 4) ? (vv << 1) : (16 + ((vv - 4) << 1))) + (grp << 3);
                a.e[2 * vv]     = d16[mn * 288 + kA];
                a.e[2 * vv + 1] = d16[mn * 288 + kA + 1];
                const int kB = (ks << 5) + ((vv < 4) ? (vv << 1) : (16 + ((vv - 4) << 1))) + (grp << 3);
                bfr.e[2 * vv]     = (_Float16)P.linW1[ng * 288 + kB];
                bfr.e[2 * vv + 1] = (_Float16)P.linW1[ng * 288 + kB + 1];
            }
            acc.v = __builtin_amdgcn_wmma_f32_16x16x32_f16(
                false, a.v, false, bfr.v, (short)0, acc.v, false, false);
        }
        #pragma unroll
        for (int r = 0; r < 8; ++r) {
            const int pr = r + (grp << 3);
            float z = acc.e[r] + P.linb1[ng];
            hmid[pr * 128 + ng] = (z >= 0.0f) ? z : 0.1f * z;
        }
    }
    __syncthreads();

    if (t < 16) {
        const int pair = p0 + t;
        if (pair < B) {
            float o = P.linb2[0];
            for (int n2 = 0; n2 < 128; ++n2) o += hmid[t * 128 + n2] * P.linW2[n2];
            out[pair] = sigmoid_f(o);
            float nn = 0.0f;
            for (int k = 0; k < 288; ++k) { float dv = d32[t * 288 + k]; nn += dv * dv; }
            out[B + pair] = sqrtf(nn);
        }
    }
}

extern "C" void kernel_launch(void* const* d_in, const int* in_sizes, int n_in,
                              void* d_out, int out_size, void* d_ws, size_t ws_size,
                              hipStream_t stream)
{
    Params P;
    int i = 0;
    P.x = (const float*)d_in[i++];
    P.y = (const float*)d_in[i++];
    for (int k = 0; k < 8; ++k) P.attW[k]  = (const float*)d_in[i++];
    for (int k = 0; k < 8; ++k) P.attb[k]  = (const float*)d_in[i++];
    for (int k = 0; k < 7; ++k) P.convW[k] = (const float*)d_in[i++];
    for (int k = 0; k < 7; ++k) P.convb[k] = (const float*)d_in[i++];
    for (int k = 0; k < 7; ++k) P.bng[k]   = (const float*)d_in[i++];
    for (int k = 0; k < 7; ++k) P.bnb[k]   = (const float*)d_in[i++];
    for (int k = 0; k < 7; ++k) P.bnm[k]   = (const float*)d_in[i++];
    for (int k = 0; k < 7; ++k) P.bnv[k]   = (const float*)d_in[i++];
    P.linW1 = (const float*)d_in[i++];
    P.linb1 = (const float*)d_in[i++];
    P.linW2 = (const float*)d_in[i++];
    P.linb2 = (const float*)d_in[i++];

    const int B = in_sizes[0] / (12 * 360);      // 2048
    float* feat = (float*)d_ws;                  // [2B][288] f32

    features_kernel<<<2 * B, BT, 0, stream>>>(P, feat, B);
    head_kernel<<<(B + 15) / 16, BT, 0, stream>>>(P, feat, (float*)d_out, B);
}
